// DAWN_25864293056823
// MI455X (gfx1250) — matrix operand order repaired and checked by reading, not compiled, not run on hardware
//
#include <hip/hip_runtime.h>
#include <math.h>
#include <stdint.h>

// ---------------------------------------------------------------------------
// Problem constants
// ---------------------------------------------------------------------------
#define BATCH     4
#define SEQ       4096
#define DMODEL    2048
#define DSPACE    128
#define NSEG      512        // neurons per segment (c / qk / v)
#define NTOT      1536
#define ROWS      (BATCH * SEQ)     // 16384
#define WAVES_PER_BLOCK 8
#define ROWS_PER_BLOCK  (WAVES_PER_BLOCK * 16)   // 128
#define KCHUNK    256        // stage-1 K staged per LDS chunk (128x256 bf16 = 64 KiB)
#define NHALF     256        // stage-2 neurons staged per LDS half (256x128 bf16 = 64 KiB)

// ---------------------------------------------------------------------------
// WMMA fragment types (CDNA5, wave32)
// ---------------------------------------------------------------------------
typedef __attribute__((ext_vector_type(16))) __bf16        bf16x16;
typedef __attribute__((ext_vector_type(8)))  float         f32x8;
typedef __attribute__((ext_vector_type(4)))  unsigned int  u32x4;

union FragAB {
    bf16x16      v;
    u32x4        q[2];
    unsigned int w[8];
};

__device__ __forceinline__ unsigned pk_bf16(float lo, float hi) {
    union { __bf16 h; unsigned short u; } a, b;
    a.h = (__bf16)lo; b.h = (__bf16)hi;
    return (unsigned)a.u | ((unsigned)b.u << 16);
}

// CDNA5 async copy: global -> LDS, 16B per lane, tracked by ASYNCcnt.
// GV addressing mode (saddr = off). LDS byte address = low 32 bits of the
// generic pointer.
__device__ __forceinline__ void async_load_b128(uint32_t lds_off, const void* gptr) {
    asm volatile("global_load_async_to_lds_b128 %0, %1, off"
                 :
                 : "v"(lds_off), "v"((uint64_t)(uintptr_t)gptr)
                 : "memory");
}
__device__ __forceinline__ void wait_async0() {
    asm volatile("s_wait_asynccnt 0x0" ::: "memory");
}

// logits/h 16x16 tile: A frags (preloaded) x B (bf16 rows, LDS or global)
__device__ __forceinline__ f32x8 mm_tile(const FragAB* A, const __bf16* nrow,
                                         int ksteps, int rowstride) {
    f32x8 c = {0.f,0.f,0.f,0.f,0.f,0.f,0.f,0.f};
    for (int qk = 0; qk < ksteps; ++qk) {
        FragAB Bf;
        Bf.q[0] = *(const u32x4*)(nrow + qk * 32);
        Bf.q[1] = *(const u32x4*)(nrow + qk * 32 + 16);
        c = __builtin_amdgcn_wmma_f32_16x16x32_bf16(false, A[qk].v, false, Bf.v,
                                                    (short)0, c, false, false);
    }
    (void)rowstride;
    return c;
}

// ---------------------------------------------------------------------------
// Prep kernels: bf16 conversion of L2-resident weights, row-normalized emb
// ---------------------------------------------------------------------------
__global__ void conv_proj_kernel(const float* __restrict__ w, __bf16* __restrict__ wbf, int n) {
    int i = blockIdx.x * blockDim.x + threadIdx.x;
    if (i < n) wbf[i] = (__bf16)w[i];
}

__global__ __launch_bounds__(128) void norm_emb_kernel(const float* __restrict__ emb,
                                                       __bf16* __restrict__ embbf) {
    __shared__ float red[128];
    const int n = blockIdx.x, i = threadIdx.x;
    float v = emb[n * DSPACE + i];
    red[i] = v * v;
    __syncthreads();
    for (int s = 64; s > 0; s >>= 1) {
        if (i < s) red[i] += red[i + s];
        __syncthreads();
    }
    float scale = 1.0f / fmaxf(sqrtf(red[0]), 1e-12f);
    embbf[n * DSPACE + i] = (__bf16)(v * scale);
}

__global__ void zero_kernel(float* __restrict__ p, int n) {
    int i = blockIdx.x * blockDim.x + threadIdx.x;
    if (i < n) p[i] = 0.f;
}

// ---------------------------------------------------------------------------
// Fused main kernel:
//   h = x @ projW^T + b     (WMMA bf16, K=2048, proj staged via async->LDS)
//   logits = h @ embN^T     (WMMA bf16, K=128, emb staged via async->LDS)
//   online 2-pass softmax per 512-segment; importance-weighted accumulation
// ---------------------------------------------------------------------------
__global__ __launch_bounds__(256) void fused_main_kernel(
    const float*  __restrict__ x,
    const float*  __restrict__ imp,
    const __bf16* __restrict__ projbf,
    const float*  __restrict__ proj_b,
    const __bf16* __restrict__ embbf,
    float*        __restrict__ dense)
{
    __shared__ __align__(16) unsigned char sbuf[KCHUNK * DSPACE * 2 * 2]; // 128 KiB staging
    __shared__ __align__(16) __bf16 sh[WAVES_PER_BLOCK][16][DSPACE];      // 32 KiB h-stash
    __shared__ float sdense[NTOT];                                        //  6 KiB

    const int tid  = threadIdx.x;
    const int wave = tid >> 5;
    const int lane = tid & 31;
    const int l16  = lane & 15;
    const int hi   = lane >> 4;            // which 16-lane half
    const int kbA  = hi * 8;               // A-frag K sub-base (16-bit A layout)
    const int kbB  = hi * 8;               // B-frag K sub-base (16-bit B layout)

    const uint32_t sb_lds = (uint32_t)(uintptr_t)&sbuf[0];
    const __bf16*  sbw    = (const __bf16*)&sbuf[0];

    for (int i = tid; i < NTOT; i += 256) sdense[i] = 0.f;

    const int rowbase = (blockIdx.x * WAVES_PER_BLOCK + wave) * 16;
    const int b       = rowbase >> 12;     // rows tile never crosses batch

    // ---------------- stage 1: h tile (16 x 128), f32 accum ----------------
    f32x8 acc[8];
#pragma unroll
    for (int t = 0; t < 8; ++t) acc[t] = (f32x8){0.f,0.f,0.f,0.f,0.f,0.f,0.f,0.f};

    const float* xrow = x + (size_t)(rowbase + l16) * DMODEL;

    for (int c = 0; c < DMODEL / KCHUNK; ++c) {
        const int kc0 = c * KCHUNK;
        // cooperative async stage of proj chunk [128 e][KCHUNK k] into LDS
        __syncthreads();   // previous chunk fully consumed before overwrite
#pragma unroll
        for (int t = 0; t < 16; ++t) {
            const int i   = tid + 256 * t;     // 4096 x 16B transfers
            const int e   = i >> 5;            // 32 transfers per 512B row
            const int kin = (i & 31) * 8;
            async_load_b128(sb_lds + (uint32_t)i * 16u,
                            projbf + (size_t)e * DMODEL + kc0 + kin);
        }
        wait_async0();
        __syncthreads();

        for (int kk = 0; kk < KCHUNK; kk += 32) {
            const int k0 = kc0 + kk;
            // A fragment: x row, f32 -> bf16 in-register (HBM traffic stays 128 MiB)
            FragAB A;
            {
                float4 f0 = *(const float4*)(xrow + k0 + kbA);
                float4 f1 = *(const float4*)(xrow + k0 + kbA + 4);
                float4 f2 = *(const float4*)(xrow + k0 + kbA + 16);
                float4 f3 = *(const float4*)(xrow + k0 + kbA + 20);
                A.w[0] = pk_bf16(f0.x, f0.y);  A.w[1] = pk_bf16(f0.z, f0.w);
                A.w[2] = pk_bf16(f1.x, f1.y);  A.w[3] = pk_bf16(f1.z, f1.w);
                A.w[4] = pk_bf16(f2.x, f2.y);  A.w[5] = pk_bf16(f2.z, f2.w);
                A.w[6] = pk_bf16(f3.x, f3.y);  A.w[7] = pk_bf16(f3.z, f3.w);
            }
#pragma unroll
            for (int t = 0; t < 8; ++t) {
                FragAB Bf;
                const __bf16* bp = sbw + (t * 16 + l16) * KCHUNK + kk + kbB;
                Bf.q[0] = *(const u32x4*)(bp);
                Bf.q[1] = *(const u32x4*)(bp + 16);
                acc[t] = __builtin_amdgcn_wmma_f32_16x16x32_bf16(false, A.v, false, Bf.v,
                                                                 (short)0, acc[t], false, false);
            }
        }
    }

    // bias add + stash h as bf16 in per-wave LDS (row-major 16x128)
#pragma unroll
    for (int t = 0; t < 8; ++t) {
        float bias = proj_b[t * 16 + l16];
#pragma unroll
        for (int j = 0; j < 8; ++j) {
            float v = acc[t][j] + bias;
            sh[wave][j + 8 * hi][t * 16 + l16] = (__bf16)v;
        }
    }

    // A fragments for stage 2 (K=128 -> 4 k-steps), via ds_load_b128
    FragAB A2[4];
    const __bf16* hrow = &sh[wave][l16][0];
#pragma unroll
    for (int qk = 0; qk < 4; ++qk) {
        A2[qk].q[0] = *(const u32x4*)(hrow + qk * 32 + kbA);
        A2[qk].q[1] = *(const u32x4*)(hrow + qk * 32 + kbA + 16);
    }

    // per-lane importance for the 8 rows this lane's C-tile half covers
    float impj[8];
#pragma unroll
    for (int j = 0; j < 8; ++j) impj[j] = imp[rowbase + j + 8 * hi];

    // ---------------- stage 2: segments c / qk / v ----------------
    for (int seg = 0; seg < 3; ++seg) {
        float m[8], z[8];
#pragma unroll
        for (int j = 0; j < 8; ++j) { m[j] = -3.0e38f; z[j] = 0.f; }

        // ---- pass 1: online (max, sum-exp), emb staged per 256-neuron half
        for (int hb = 0; hb < 2; ++hb) {
            __syncthreads();   // previous contents consumed
            const __bf16* src = embbf + (size_t)(seg * NSEG + hb * NHALF) * DSPACE;
#pragma unroll
            for (int t = 0; t < 16; ++t) {
                const int i = tid + 256 * t;   // 64 KiB contiguous
                async_load_b128(sb_lds + (uint32_t)i * 16u, src + (size_t)i * 8);
            }
            wait_async0();
            __syncthreads();

            for (int tile = 0; tile < NHALF / 16; ++tile) {
                const __bf16* nrow = sbw + (tile * 16 + l16) * DSPACE + kbB;
                f32x8 cst = mm_tile(A2, nrow, 4, DSPACE);
#pragma unroll
                for (int j = 0; j < 8; ++j) {
                    float v  = cst[j];
                    float mn = fmaxf(m[j], v);
                    z[j] = z[j] * __expf(m[j] - mn) + __expf(v - mn);
                    m[j] = mn;
                }
            }
        }

        // combine partial stats across the 16-lane column group (stay inside half!)
#pragma unroll
        for (int off = 1; off < 16; off <<= 1) {
#pragma unroll
            for (int j = 0; j < 8; ++j) {
                float mo = __shfl_xor(m[j], off, 16);
                float zo = __shfl_xor(z[j], off, 16);
                float mn = fmaxf(m[j], mo);
                z[j] = z[j] * __expf(m[j] - mn) + zo * __expf(mo - mn);
                m[j] = mn;
            }
        }
        float fac[8];
#pragma unroll
        for (int j = 0; j < 8; ++j) fac[j] = impj[j] / z[j];

        // ---- pass 2: recompute logits, accumulate weighted probs
        for (int hb = 0; hb < 2; ++hb) {
            __syncthreads();
            const __bf16* src = embbf + (size_t)(seg * NSEG + hb * NHALF) * DSPACE;
#pragma unroll
            for (int t = 0; t < 16; ++t) {
                const int i = tid + 256 * t;
                async_load_b128(sb_lds + (uint32_t)i * 16u, src + (size_t)i * 8);
            }
            wait_async0();
            __syncthreads();

            for (int tile = 0; tile < NHALF / 16; ++tile) {
                const __bf16* nrow = sbw + (tile * 16 + l16) * DSPACE + kbB;
                f32x8 cst = mm_tile(A2, nrow, 4, DSPACE);
                float part = 0.f;
#pragma unroll
                for (int j = 0; j < 8; ++j)
                    part += __expf(cst[j] - m[j]) * fac[j];
                atomicAdd(&sdense[seg * NSEG + hb * NHALF + tile * 16 + l16], part);
            }
        }
    }

    __syncthreads();
    for (int i = tid; i < NTOT; i += 256) {
        int seg = i >> 9, nl = i & (NSEG - 1);
        atomicAdd(&dense[seg * (BATCH * NSEG) + b * NSEG + nl], sdense[i]);
    }
}

// ---------------------------------------------------------------------------
// Top-k sparsify + renormalize.  grid = 12 blocks: kind (c/qk/v) x batch.
// ---------------------------------------------------------------------------
__global__ __launch_bounds__(256) void topk_kernel(const float* __restrict__ dense,
                                                   float* __restrict__ out)
{
    __shared__ float sv[NSEG];
    __shared__ float wv[NSEG];
    __shared__ int   sel[NSEG];
    __shared__ float rmax[256];
    __shared__ int   ridx[256];
    __shared__ float ssum;

    const int kind = blockIdx.x >> 2;      // 0=c, 1=qk, 2=v
    const int b    = blockIdx.x & 3;
    const int K    = (kind == 0) ? 8 : ((kind == 1) ? 4 : 6);
    const float* row = dense + kind * (BATCH * NSEG) + b * NSEG;
    const int tid = threadIdx.x;

    for (int i = tid; i < NSEG; i += 256) { sv[i] = row[i]; wv[i] = row[i]; sel[i] = 0; }
    if (tid == 0) ssum = 0.f;
    __syncthreads();

    for (int it = 0; it < K; ++it) {
        float v0 = wv[tid], v1 = wv[tid + 256];
        float mv; int mi;
        if (v1 > v0) { mv = v1; mi = tid + 256; } else { mv = v0; mi = tid; }
        rmax[tid] = mv; ridx[tid] = mi;
        __syncthreads();
        for (int s = 128; s > 0; s >>= 1) {
            if (tid < s && rmax[tid + s] > rmax[tid]) {
                rmax[tid] = rmax[tid + s]; ridx[tid] = ridx[tid + s];
            }
            __syncthreads();
        }
        if (tid == 0) {
            int w = ridx[0];
            sel[w] = 1; ssum += sv[w]; wv[w] = -3.0e38f;
        }
        __syncthreads();
    }

    float inv = 1.0f / (ssum + 1e-8f);
    for (int i = tid; i < NSEG; i += 256) {
        float o = sel[i] ? sv[i] * inv : 0.f;
        if (kind == 0)      { out[0 * 2048 + b * NSEG + i] = o; }
        else if (kind == 1) { out[1 * 2048 + b * NSEG + i] = o;
                              out[2 * 2048 + b * NSEG + i] = o; }
        else                { out[3 * 2048 + b * NSEG + i] = o; }
    }
}

// ---------------------------------------------------------------------------
// Launch
// ---------------------------------------------------------------------------
extern "C" void kernel_launch(void* const* d_in, const int* in_sizes, int n_in,
                              void* d_out, int out_size, void* d_ws, size_t ws_size,
                              hipStream_t stream) {
    const float* x      = (const float*)d_in[0];
    const float* imp    = (const float*)d_in[1];
    const float* proj_w = (const float*)d_in[2];
    const float* proj_b = (const float*)d_in[3];
    const float* emb    = (const float*)d_in[4];
    float* out = (float*)d_out;

    char* ws = (char*)d_ws;
    __bf16* projbf = (__bf16*)ws;                              // 128*2048 bf16 = 512 KiB
    __bf16* embbf  = (__bf16*)(ws + 524288);                   // 1536*128 bf16 = 384 KiB
    float*  dense  = (float*)(ws + 524288 + 393216);           // 3*4*512 f32  =  24 KiB

    conv_proj_kernel<<<(DSPACE * DMODEL + 255) / 256, 256, 0, stream>>>(proj_w, projbf,
                                                                        DSPACE * DMODEL);
    norm_emb_kernel<<<NTOT, 128, 0, stream>>>(emb, embbf);
    zero_kernel<<<(3 * BATCH * NSEG + 255) / 256, 256, 0, stream>>>(dense, 3 * BATCH * NSEG);
    fused_main_kernel<<<ROWS / ROWS_PER_BLOCK, 256, 0, stream>>>(x, imp, projbf, proj_b,
                                                                 embbf, dense);
    topk_kernel<<<12, 256, 0, stream>>>(dense, out);
}
